// CoAttentionModule_59511066853740
// MI455X (gfx1250) — matrix operand order repaired and checked by reading, not compiled, not run on hardware
//
#include <hip/hip_runtime.h>

#define B_  4
#define C_  2048
#define HW_ 2304
#define HC_ 256

typedef __attribute__((ext_vector_type(16))) __bf16 bf16x16;
typedef __attribute__((ext_vector_type(8)))  float  floatx8;
typedef __attribute__((ext_vector_type(4)))  unsigned int uint32x4;
typedef __attribute__((ext_vector_type(8)))  int int32x8;
typedef __attribute__((ext_vector_type(4)))  int int32x4;

#if defined(__has_builtin)
#if __has_builtin(__builtin_amdgcn_tensor_load_to_lds) && __has_builtin(__builtin_amdgcn_s_wait_tensorcnt)
#define HAVE_TDM 1
#endif
#endif
#ifndef HAVE_TDM
#define HAVE_TDM 0
#endif

#if HAVE_TDM
#warning "CDNA5 probe: tensor_load_to_lds builtin AVAILABLE - TDM double-buffered path enabled"
#else
#warning "CDNA5 probe: tensor_load_to_lds builtin NOT available - manual LDS copy fallback"
#endif

union FragBF { uint4 q[2]; bf16x16 v; };

__device__ __forceinline__ unsigned short f32_to_bf16(float f) {
    unsigned int u = __float_as_uint(f);
    return (unsigned short)((u + 0x7FFFu + ((u >> 16) & 1u)) >> 16);
}
__device__ __forceinline__ unsigned pack_bf16x2(float a, float b) {
    return (unsigned)f32_to_bf16(a) | ((unsigned)f32_to_bf16(b) << 16);
}

// ---- wave-level 32x64 WMMA step: 2 A frags (16x32) x 4 B frags (32x16) = 8 WMMA ----
__device__ __forceinline__ void wave_mma_step2(const unsigned short* lsA,
                                               const unsigned short* lsB,
                                               int wm, int wn, int lr, int hi,
                                               floatx8 acc[2][4]) {
    FragBF fa[2];
    const int off = hi * 8;             // lanes 0-15: K 0..7,16..23 ; 16-31: K 8..15,24..31
#pragma unroll
    for (int r = 0; r < 2; ++r) {
        const int row = wm * 32 + r * 16 + lr;
        fa[r].q[0] = *reinterpret_cast<const uint4*>(&lsA[row * 32 + off]);
        fa[r].q[1] = *reinterpret_cast<const uint4*>(&lsA[row * 32 + 16 + off]);
    }
    const int kh = hi * 8;              // lanes 0-15: K 0..7,16..23 ; 16-31: K 8..15,24..31
#pragma unroll
    for (int f = 0; f < 4; ++f) {
        FragBF fb;
        const int n = wn * 64 + f * 16 + lr;
        fb.q[0] = *reinterpret_cast<const uint4*>(&lsB[n * 32 + kh]);
        fb.q[1] = *reinterpret_cast<const uint4*>(&lsB[n * 32 + kh + 16]);
#pragma unroll
        for (int r = 0; r < 2; ++r)
            acc[r][f] = __builtin_amdgcn_wmma_f32_16x16x32_bf16(
                false, fa[r].v, false, fb.v, (short)0, acc[r][f], false, false);
    }
}

#if HAVE_TDM
// Issue one TDM 2D tile load: 256 rows x 32 bf16 elems, row stride 4608 elems,
// packed contiguously into LDS (64B per row). D# per CDNA5 ISA ch.8.
__device__ __forceinline__ void tdm_issue(const unsigned short* lds_dst,
                                          const unsigned short* Pr,
                                          int Nb, int k0) {
    unsigned lds_base = (unsigned)(size_t)lds_dst;
    unsigned long long ga = (unsigned long long)(size_t)(&Pr[(size_t)Nb * (HW_ * 2) + k0]);
    uint32x4 g0;
    g0[0] = 1u;                                                // count=1 (user descriptor)
    g0[1] = lds_base;                                          // lds_addr
    g0[2] = (unsigned)(ga & 0xFFFFFFFFu);                      // global_addr[31:0]
    g0[3] = (unsigned)((ga >> 32) & 0x01FFFFFFu) | 0x80000000u; // addr[56:32] | type=2
    int32x8 g1;
    g1[0] = 0x10000;                                           // data_size=1 (2 bytes)
    g1[1] = (int)((4608u & 0xFFFFu) << 16);                    // tensor_dim0[15:0]
    g1[2] = (int)((4608u >> 16) | ((4096u & 0xFFFFu) << 16));  // td0 hi | td1 lo
    g1[3] = (int)((4096u >> 16) | (32u << 16));                // td1 hi | tile_dim0=32
    g1[4] = (int)(256u);                                       // tile_dim1=256 | tile_dim2=0
    g1[5] = (int)4608;                                         // tensor_dim0_stride[31:0]
    g1[6] = 0;
    g1[7] = 0;
    int32x4 z4 = {0, 0, 0, 0};
#if __clang_major__ >= 23
    int32x8 z8 = {0, 0, 0, 0, 0, 0, 0, 0};
    __builtin_amdgcn_tensor_load_to_lds(g0, g1, z4, z4, z8, 0);
#else
    __builtin_amdgcn_tensor_load_to_lds(g0, g1, z4, z4, 0);
#endif
}
#endif

// ---------------- copy originals into both output halves ----------------
__global__ __launch_bounds__(256) void copy_orig_kernel(const float4* __restrict__ L,
                                                        const float4* __restrict__ R,
                                                        float4* __restrict__ Out) {
    const size_t per_batch = (size_t)C_ * HW_ / 4;
    size_t gid = (size_t)blockIdx.x * 256 + threadIdx.x;
    size_t b = gid / per_batch, r = gid % per_batch;
    size_t dst = b * (2 * per_batch) + r;
    Out[dst] = L[gid];
    Out[(size_t)B_ * 2 * per_batch + dst] = R[gid];
}

// ---------------- projections: P[p][b] = bf16(W * X + bias) ----------------
__global__ __launch_bounds__(256) void proj_kernel(const float* __restrict__ L,
                                                   const float* __restrict__ R,
                                                   const float* __restrict__ Wq,
                                                   const float* __restrict__ bq,
                                                   const float* __restrict__ Wk,
                                                   const float* __restrict__ bk,
                                                   unsigned short* __restrict__ P) {
    __shared__ alignas(16) unsigned short lsA[64 * 32];
    __shared__ alignas(16) unsigned short lsB[256 * 32];

    const int t = threadIdx.x;
    const int z = blockIdx.z;
    const int p = z >> 2, b = z & 3;        // p: 0=Wq*L 1=Wk*R 2=Wq*R 3=Wk*L
    const float* Wm   = (p & 1) ? Wk : Wq;
    const float* bias = (p & 1) ? bk : bq;
    const float* X    = ((p == 1) || (p == 2)) ? R : L;
    X += (size_t)b * C_ * HW_;
    unsigned short* Pout = P + (size_t)(p * B_ + b) * HC_ * HW_;

    const int Nb = blockIdx.x * 256;
    const int Mb = blockIdx.y * 64;
    const int lane = t & 31, w = t >> 5;
    const int wm = w & 1, wn = w >> 1;
    const int hi = lane >> 4, lr = lane & 15;

    const int sa_m = t >> 2, sa_k = (t & 3) * 8;
    const int sb_k = t >> 3, sb_n = (t & 7) * 32;

    floatx8 acc[2][4] = {};

    for (int k0 = 0; k0 < C_; k0 += 32) {
        __syncthreads();
        {   // stage A (f32 -> bf16), row-major [m][k], 8 k per thread
            const float* src = &Wm[(size_t)(Mb + sa_m) * C_ + k0 + sa_k];
            float4 f0 = *reinterpret_cast<const float4*>(src);
            float4 f1 = *reinterpret_cast<const float4*>(src + 4);
            if (k0 + 32 < C_) __builtin_prefetch(src + 32, 0, 0);
            uint4 pk;
            pk.x = pack_bf16x2(f0.x, f0.y); pk.y = pack_bf16x2(f0.z, f0.w);
            pk.z = pack_bf16x2(f1.x, f1.y); pk.w = pack_bf16x2(f1.z, f1.w);
            *reinterpret_cast<uint4*>(&lsA[sa_m * 32 + sa_k]) = pk;
        }
        {   // stage B transposed: lsB[n][k] = bf16(X[k][n])
            const float* src = &X[(size_t)(k0 + sb_k) * HW_ + Nb + sb_n];
#pragma unroll
            for (int j4 = 0; j4 < 8; ++j4) {
                float4 f = *reinterpret_cast<const float4*>(src + j4 * 4);
                lsB[(sb_n + j4 * 4 + 0) * 32 + sb_k] = f32_to_bf16(f.x);
                lsB[(sb_n + j4 * 4 + 1) * 32 + sb_k] = f32_to_bf16(f.y);
                lsB[(sb_n + j4 * 4 + 2) * 32 + sb_k] = f32_to_bf16(f.z);
                lsB[(sb_n + j4 * 4 + 3) * 32 + sb_k] = f32_to_bf16(f.w);
            }
        }
        __syncthreads();
        wave_mma_step2(lsA, lsB, wm, wn, lr, hi, acc);
    }

#pragma unroll
    for (int r = 0; r < 2; ++r)
#pragma unroll
    for (int v = 0; v < 8; ++v) {
        const int m = Mb + wm * 32 + r * 16 + v + 8 * hi;
        const float bv = bias[m];
#pragma unroll
        for (int f = 0; f < 4; ++f) {
            const int n = Nb + wn * 64 + f * 16 + lr;
            Pout[(size_t)m * HW_ + n] = f32_to_bf16(acc[r][f][v] + bv);
        }
    }
}

// ---------------- scores: S[i][j] = sum_hc Q[hc][i]*K[hc][j] (f32) ----------------
__global__ __launch_bounds__(256) void scores_kernel(const unsigned short* __restrict__ Q,
                                                     const unsigned short* __restrict__ K,
                                                     float* __restrict__ S) {
    __shared__ alignas(16) unsigned short lsA[64 * 32];
    __shared__ alignas(16) unsigned short lsB[256 * 32];

    const int t = threadIdx.x, b = blockIdx.z;
    Q += (size_t)b * HC_ * HW_;
    K += (size_t)b * HC_ * HW_;
    S += (size_t)b * HW_ * HW_;

    const int Nb = blockIdx.x * 256;
    const int Mb = blockIdx.y * 64;
    const int lane = t & 31, w = t >> 5;
    const int wm = w & 1, wn = w >> 1;
    const int hi = lane >> 4, lr = lane & 15;

    const int sa_k = t >> 3, sa_m = (t & 7) * 8;
    const int sb_k = t >> 3, sb_n = (t & 7) * 32;

    floatx8 acc[2][4] = {};

    for (int k0 = 0; k0 < HC_; k0 += 32) {
        __syncthreads();
        {   // A transposed: lsA[m][k] = Q[k][Mb+m], 8 m per thread
            uint4 d = *reinterpret_cast<const uint4*>(&Q[(size_t)(k0 + sa_k) * HW_ + Mb + sa_m]);
            lsA[(sa_m + 0) * 32 + sa_k] = (unsigned short)(d.x & 0xFFFFu);
            lsA[(sa_m + 1) * 32 + sa_k] = (unsigned short)(d.x >> 16);
            lsA[(sa_m + 2) * 32 + sa_k] = (unsigned short)(d.y & 0xFFFFu);
            lsA[(sa_m + 3) * 32 + sa_k] = (unsigned short)(d.y >> 16);
            lsA[(sa_m + 4) * 32 + sa_k] = (unsigned short)(d.z & 0xFFFFu);
            lsA[(sa_m + 5) * 32 + sa_k] = (unsigned short)(d.z >> 16);
            lsA[(sa_m + 6) * 32 + sa_k] = (unsigned short)(d.w & 0xFFFFu);
            lsA[(sa_m + 7) * 32 + sa_k] = (unsigned short)(d.w >> 16);
        }
        {   // B transposed: lsB[n][k] = K[k][Nb+n]
            const unsigned short* src = &K[(size_t)(k0 + sb_k) * HW_ + Nb + sb_n];
#pragma unroll
            for (int j = 0; j < 4; ++j) {
                uint4 d = *reinterpret_cast<const uint4*>(src + j * 8);
                const int nb = sb_n + j * 8;
                lsB[(nb + 0) * 32 + sb_k] = (unsigned short)(d.x & 0xFFFFu);
                lsB[(nb + 1) * 32 + sb_k] = (unsigned short)(d.x >> 16);
                lsB[(nb + 2) * 32 + sb_k] = (unsigned short)(d.y & 0xFFFFu);
                lsB[(nb + 3) * 32 + sb_k] = (unsigned short)(d.y >> 16);
                lsB[(nb + 4) * 32 + sb_k] = (unsigned short)(d.z & 0xFFFFu);
                lsB[(nb + 5) * 32 + sb_k] = (unsigned short)(d.z >> 16);
                lsB[(nb + 6) * 32 + sb_k] = (unsigned short)(d.w & 0xFFFFu);
                lsB[(nb + 7) * 32 + sb_k] = (unsigned short)(d.w >> 16);
            }
        }
        __syncthreads();
        wave_mma_step2(lsA, lsB, wm, wn, lr, hi, acc);
    }

#pragma unroll
    for (int r = 0; r < 2; ++r)
#pragma unroll
    for (int v = 0; v < 8; ++v) {
        const int i = Mb + wm * 32 + r * 16 + v + 8 * hi;
#pragma unroll
        for (int f = 0; f < 4; ++f) {
            const int j = Nb + wn * 64 + f * 16 + lr;
            S[(size_t)i * HW_ + j] = acc[r][f][v];
        }
    }
}

// ---------------- softmax rows; write bf16 probs compacted in-place ----------------
__global__ __launch_bounds__(256) void softmax_kernel(float* __restrict__ S) {
    __shared__ float red[256];
    const size_t row = blockIdx.x;
    float* rowp = S + row * (size_t)HW_;
    const int t = threadIdx.x;

    float v[9];
    float m = -__builtin_inff();
#pragma unroll
    for (int e = 0; e < 9; ++e) { v[e] = rowp[t + e * 256]; m = fmaxf(m, v[e]); }
    red[t] = m; __syncthreads();
    for (int s = 128; s > 0; s >>= 1) { if (t < s) red[t] = fmaxf(red[t], red[t + s]); __syncthreads(); }
    m = red[0]; __syncthreads();

    float sum = 0.f;
#pragma unroll
    for (int e = 0; e < 9; ++e) { v[e] = __expf(v[e] - m); sum += v[e]; }
    red[t] = sum; __syncthreads();
    for (int s = 128; s > 0; s >>= 1) { if (t < s) red[t] += red[t + s]; __syncthreads(); }
    const float inv = 1.0f / red[0];
    __syncthreads();

    unsigned short* outp = reinterpret_cast<unsigned short*>(rowp);
#pragma unroll
    for (int e = 0; e < 9; ++e) outp[t + e * 256] = f32_to_bf16(v[e] * inv);
}

// ---------------- out = V * A^T into "weighted" channel block ----------------
__global__ __launch_bounds__(256) void outgemm_kernel(const float* __restrict__ V,
                                                      const unsigned short* __restrict__ Pr,
                                                      float* __restrict__ Out) {
    __shared__ alignas(16) unsigned short lsA[64 * 32];
#if HAVE_TDM
    __shared__ alignas(16) unsigned short lsB[2][256 * 32];   // ping-pong TDM destination
#else
    __shared__ alignas(16) unsigned short lsB1[256 * 32];
#endif

    const int t = threadIdx.x, b = blockIdx.z;
    V  += (size_t)b * C_ * HW_;
    Pr += (size_t)b * HW_ * (HW_ * 2);      // bf16 rows, stride = 2*HW ushorts (in-place f32 buf)

    const int Nb = blockIdx.x * 256;        // i (query pos)
    const int Mb = blockIdx.y * 64;         // c (channel)
    const int lane = t & 31, w = t >> 5;
    const int wm = w & 1, wn = w >> 1;
    const int hi = lane >> 4, lr = lane & 15;

    const int sa_m = t >> 2, sa_k = (t & 3) * 8;

    floatx8 acc[2][4] = {};

#if HAVE_TDM
    if (t == 0) tdm_issue(&lsB[0][0], Pr, Nb, 0);   // prime the pipeline
    int buf = 0;
    for (int k0 = 0; k0 < HW_; k0 += 32) {
        __syncthreads();                    // prior reads of lsA and lsB[buf^1] complete
        {   // stage A (f32 V -> bf16): lsA[m][k] = V[Mb+m][k0+k], 8 k per thread
            const float* src = &V[(size_t)(Mb + sa_m) * HW_ + k0 + sa_k];
            float4 f0 = *reinterpret_cast<const float4*>(src);
            float4 f1 = *reinterpret_cast<const float4*>(src + 4);
            if (k0 + 32 < HW_) __builtin_prefetch(src + 32, 0, 0);
            uint4 pk;
            pk.x = pack_bf16x2(f0.x, f0.y); pk.y = pack_bf16x2(f0.z, f0.w);
            pk.z = pack_bf16x2(f1.x, f1.y); pk.w = pack_bf16x2(f1.z, f1.w);
            *reinterpret_cast<uint4*>(&lsA[sa_m * 32 + sa_k]) = pk;
        }
        const bool has_next = (k0 + 32) < HW_;
        if (has_next && t == 0) tdm_issue(&lsB[buf ^ 1][0], Pr, Nb, k0 + 32);
        // TDM ops from one wave retire in order: <=1 outstanding means the
        // current tile has landed while the just-issued next tile stays in flight.
        if (has_next) __builtin_amdgcn_s_wait_tensorcnt(1);
        else          __builtin_amdgcn_s_wait_tensorcnt(0);
        __syncthreads();
        wave_mma_step2(lsA, &lsB[buf][0], wm, wn, lr, hi, acc);
        buf ^= 1;
    }
#else
    for (int k0 = 0; k0 < HW_; k0 += 32) {
        __syncthreads();
        {   // stage A (f32 V -> bf16)
            const float* src = &V[(size_t)(Mb + sa_m) * HW_ + k0 + sa_k];
            float4 f0 = *reinterpret_cast<const float4*>(src);
            float4 f1 = *reinterpret_cast<const float4*>(src + 4);
            if (k0 + 32 < HW_) __builtin_prefetch(src + 32, 0, 0);
            uint4 pk;
            pk.x = pack_bf16x2(f0.x, f0.y); pk.y = pack_bf16x2(f0.z, f0.w);
            pk.z = pack_bf16x2(f1.x, f1.y); pk.w = pack_bf16x2(f1.z, f1.w);
            *reinterpret_cast<uint4*>(&lsA[sa_m * 32 + sa_k]) = pk;
        }
        {   // stage B: lsB1[n][k] = probs[Nb+n][k0+k]  (k-contiguous -> b128 copy)
            const uint4* src = reinterpret_cast<const uint4*>(&Pr[(size_t)(Nb + t) * (HW_ * 2) + k0]);
            uint4* dst = reinterpret_cast<uint4*>(&lsB1[t * 32]);
            dst[0] = src[0]; dst[1] = src[1]; dst[2] = src[2]; dst[3] = src[3];
        }
        __syncthreads();
        wave_mma_step2(lsA, lsB1, wm, wn, lr, hi, acc);
    }
#endif

#pragma unroll
    for (int r = 0; r < 2; ++r)
#pragma unroll
    for (int v = 0; v < 8; ++v) {
        const int c = Mb + wm * 32 + r * 16 + v + 8 * hi;
#pragma unroll
        for (int f = 0; f < 4; ++f) {
            const int i = Nb + wn * 64 + f * 16 + lr;
            Out[((size_t)(b * 2 * C_ + C_ + c)) * HW_ + i] = acc[r][f][v];
        }
    }
}

extern "C" void kernel_launch(void* const* d_in, const int* in_sizes, int n_in,
                              void* d_out, int out_size, void* d_ws, size_t ws_size,
                              hipStream_t stream) {
    const float* L  = (const float*)d_in[0];
    const float* R  = (const float*)d_in[1];
    const float* Wq = (const float*)d_in[2];
    const float* bq = (const float*)d_in[3];
    const float* Wk = (const float*)d_in[4];
    const float* bk = (const float*)d_in[5];
    float* out = (float*)d_out;

    // ws layout: P (4 projections, bf16) | Sbuf (one direction of logits/probs, f32)
    unsigned short* P = (unsigned short*)d_ws;
    const size_t P_bytes = (size_t)4 * B_ * HC_ * HW_ * sizeof(unsigned short);
    float* Sbuf = (float*)((char*)d_ws + P_bytes);

    copy_orig_kernel<<<dim3((B_ * C_ * HW_ / 4) / 256), dim3(256), 0, stream>>>(
        (const float4*)L, (const float4*)R, (float4*)out);

    proj_kernel<<<dim3(HW_ / 256, HC_ / 64, 4 * B_), dim3(256), 0, stream>>>(
        L, R, Wq, bq, Wk, bk, P);

    for (int dir = 0; dir < 2; ++dir) {
        const unsigned short* Qp = P + (size_t)((dir == 0 ? 0 : 2) * B_) * HC_ * HW_;
        const unsigned short* Kp = P + (size_t)((dir == 0 ? 1 : 3) * B_) * HC_ * HW_;

        scores_kernel<<<dim3(HW_ / 256, HW_ / 64, B_), dim3(256), 0, stream>>>(Qp, Kp, Sbuf);
        softmax_kernel<<<dim3(B_ * HW_), dim3(256), 0, stream>>>(Sbuf);

        const float* V = (dir == 0) ? R : L;
        float* outbase = out + (dir == 0 ? (size_t)B_ * 2 * C_ * HW_ : 0);
        outgemm_kernel<<<dim3(HW_ / 256, C_ / 64, B_), dim3(256), 0, stream>>>(
            V, (const unsigned short*)Sbuf, outbase);
    }
}
